// LSHAttention_82154134438588
// MI455X (gfx1250) — compile-verified, matrix operand order checked by reading, not run on hardware
//
#include <hip/hip_runtime.h>
#include <math.h>

// ---------------------------------------------------------------------------
// LSH attention (Reformer-style) for MI455X / gfx1250, wave32 + WMMA f16.
// B=2 H=8 L=2048 D=64 R=4 bucket=64  -> bh=16, nb2=16, chunk=128, keys=256
// ---------------------------------------------------------------------------

typedef __attribute__((ext_vector_type(16))) _Float16 v16h;
typedef __attribute__((ext_vector_type(8)))  float    v8f;
typedef __attribute__((ext_vector_type(4)))  int      v4i;

#define BHN   16
#define LSEQ  2048
#define DIM   64
#define RR    4
#define NB2   16
#define CH    128          // 2 * bucket (queries per chunk)
#define KW    256          // 4 * bucket (keys per chunk, with look-back)

// ---- gfx1250 async global->LDS path (ASYNCcnt), with safe fallback ---------
#if defined(__AMDGCN__) && __has_builtin(__builtin_amdgcn_global_load_async_to_lds_b128) && \
    __has_builtin(__builtin_amdgcn_s_wait_asynccnt)
#define USE_ASYNC 1
#else
#define USE_ASYNC 0
#endif

#if USE_ASYNC
__device__ __forceinline__ void async_copy16(const void* g, void* l) {
  // builtin takes (addrspace(1) int4*, addrspace(3) int4*, imm offset, imm cpol);
  // low 32 bits of a generic LDS pointer are the LDS byte offset.
  __builtin_amdgcn_global_load_async_to_lds_b128(
      (__attribute__((address_space(1))) v4i*)(unsigned long long)(size_t)g,
      (__attribute__((address_space(3))) v4i*)(unsigned)(size_t)l,
      0, 0);
}
#define ASYNC_WAIT() __builtin_amdgcn_s_wait_asynccnt(0)
#else
#define ASYNC_WAIT() ((void)0)
#endif

// copy one 128-byte row (64 halfs) global -> LDS
__device__ __forceinline__ void copy_row64h(const _Float16* __restrict__ src,
                                            _Float16* dst) {
#if USE_ASYNC
#pragma unroll
  for (int w = 0; w < 8; ++w) async_copy16(src + w * 8, dst + w * 8);
#else
  const uint4* s4 = (const uint4*)src;
  uint4* d4 = (uint4*)dst;
#pragma unroll
  for (int w = 0; w < 8; ++w) d4[w] = s4[w];
#endif
}

// ---- WMMA fragment loaders (CDNA5 ISA 7.12.2 16-bit striping) --------------
// A / B fragments: lane L -> row/col = L&15, K half-group = L>>4.
// 16 packed f16 per lane; half h -> K = (h>>3)*16 + (L>>4)*8 + (h&7).

__device__ __forceinline__ v16h frag_rowmajor(const _Float16* base, int row0,
                                              int k0, int stride, int lane) {
  const int rsel = row0 + (lane & 15);
  const int c    = (lane >> 4) & 1;
  const _Float16* p = base + rsel * stride + k0;
  v16h f;
#pragma unroll
  for (int h = 0; h < 16; ++h) {
    int k = ((h >> 3) << 4) + (c << 3) + (h & 7);
    f[h] = p[k];
  }
  return f;
}

__device__ __forceinline__ v16h frag_kmajor(const _Float16* base, int n0,
                                            int k0, int stride, int lane) {
  const int n = n0 + (lane & 15);
  const int c = (lane >> 4) & 1;
  v16h f;
#pragma unroll
  for (int h = 0; h < 16; ++h) {
    int k = k0 + ((h >> 3) << 4) + (c << 3) + (h & 7);
    f[h] = base[k * stride + n];
  }
  return f;
}

// ---- 1) normalize rand_matrix columns over d -------------------------------
// rm layout (bh, d, R, nb2); block = (bh, r, k); thread = d.
__global__ __launch_bounds__(64) void norm_rm_kernel(const float* __restrict__ rm,
                                                     float* __restrict__ rmn) {
  __shared__ float red[64];
  int bid = blockIdx.x;
  int k = bid & 15, r = (bid >> 4) & 3, bh = bid >> 6;
  int d = threadIdx.x;
  float x = rm[(size_t)bh * 4096 + (size_t)d * 64 + r * 16 + k];
  red[d] = x * x;
  __syncthreads();
  for (int s = 32; s > 0; s >>= 1) { if (d < s) red[d] += red[d + s]; __syncthreads(); }
  float rn = rsqrtf(red[0] + 1e-30f);
  rmn[((size_t)bh * 64 + (r * 16 + k)) * 64 + d] = x * rn;
}

// ---- 2) normalize q rows, f16 copies of q & v, proj, hashes ----------------
__global__ __launch_bounds__(64) void norm_q_hash_kernel(
    const float* __restrict__ q, const float* __restrict__ v,
    const float* __restrict__ rmn, _Float16* __restrict__ qh,
    _Float16* __restrict__ vh, int* __restrict__ hashes) {
  __shared__ float row[64];
  __shared__ float proj[64];
  __shared__ float red[64];
  int bid = blockIdx.x;
  int bh = bid >> 11, t = bid & (LSEQ - 1);
  int d = threadIdx.x;
  size_t base = ((size_t)bh * LSEQ + t) * DIM;
  float qv = q[base + d];
  red[d] = qv * qv;
  __syncthreads();
  for (int s = 32; s > 0; s >>= 1) { if (d < s) red[d] += red[d + s]; __syncthreads(); }
  float qn = qv * rsqrtf(red[0] + 1e-30f);
  row[d] = qn;
  qh[base + d] = (_Float16)qn;
  vh[base + d] = (_Float16)v[base + d];
  __syncthreads();
  // proj[rk] = sum_d qn[d] * rmn[bh][rk][d]   (rk = r*16+k, thread = rk)
  const float* rmrow = rmn + ((size_t)bh * 64 + d) * 64;
  float acc = 0.f;
#pragma unroll 8
  for (int dd = 0; dd < 64; ++dd) acc += row[dd] * rmrow[dd];
  proj[d] = acc;
  __syncthreads();
  if (d < RR) {
    int r = d;
    float best = -3.4e38f; int bi = 0;
    for (int k = 0; k < 16; ++k) { float p = proj[r * 16 + k]; if (p > best) { best = p; bi = k; } }
    for (int k = 0; k < 16; ++k) { float p = -proj[r * 16 + k]; if (p > best) { best = p; bi = 16 + k; } }
    hashes[(bh * RR + r) * LSEQ + t] = bi;
  }
}

// ---- 3) stable counting sort per (bh, r): 32 hash bins ---------------------
__global__ __launch_bounds__(256) void sort_kernel(
    const int* __restrict__ hashes, int* __restrict__ hash_idx,
    int* __restrict__ sorted_h, int* __restrict__ orig_idx) {
  __shared__ int hist[32];
  __shared__ int off[32];
  __shared__ int hl[LSEQ];
  int bid = blockIdx.x;            // bh*4 + r
  int tid = threadIdx.x;
  const int* h = hashes + (size_t)bid * LSEQ;
  if (tid < 32) hist[tid] = 0;
  __syncthreads();
  for (int t = tid; t < LSEQ; t += 256) { int hv = h[t]; hl[t] = hv; atomicAdd(&hist[hv], 1); }
  __syncthreads();
  if (tid == 0) {
    int s = 0;
    for (int b = 0; b < 32; ++b) { off[b] = s; s += hist[b]; }
    for (int t = 0; t < LSEQ; ++t) {   // serial stable scatter
      int hv = hl[t];
      int p = off[hv]++;
      hash_idx[(size_t)bid * LSEQ + p] = t;
      sorted_h[(size_t)bid * LSEQ + p] = hv;
      orig_idx[(size_t)bid * LSEQ + t] = p;
    }
  }
}

// ---- 4) scores: S = Q*K^T/8 via WMMA, masked, scattered to token order -----
// block = (bh, n, r); 256 threads = 8 waves; each wave owns row-tile ti=wave.
__global__ __launch_bounds__(256) void scores_kernel(
    const _Float16* __restrict__ qh, const int* __restrict__ hash_idx,
    const int* __restrict__ sorted_h, int* __restrict__ kidx_g,
    float* __restrict__ scores_g) {
  __shared__ _Float16 Qs[CH][72];
  __shared__ _Float16 Ks[KW][72];
  __shared__ int qi_s[CH], sh_s[CH], ki_s[KW], kh_s[KW];
  int blk = blockIdx.x;
  int r = blk & 3, n = (blk >> 2) & 15, bh = blk >> 6;
  int tid = threadIdx.x;
  int prevn = (n + NB2 - 1) & 15;
  const int* hidx = hash_idx + (size_t)(bh * RR + r) * LSEQ;
  const int* shh  = sorted_h + (size_t)(bh * RR + r) * LSEQ;
  if (tid < CH) {
    int i = tid;
    int tok = hidx[n * CH + i];
    qi_s[i] = tok;
    sh_s[i] = shh[n * CH + i];
    copy_row64h(qh + ((size_t)bh * LSEQ + tok) * DIM, &Qs[i][0]);
  }
  {
    int j = tid;                          // 0..255 key rows
    int chunk = (j < CH) ? prevn : n;     // look-back: prev chunk ++ this chunk
    int pos = chunk * CH + (j & (CH - 1));
    int tok = hidx[pos];
    ki_s[j] = tok;
    kh_s[j] = shh[pos];
    kidx_g[((size_t)(bh * RR + r) * NB2 + n) * KW + j] = tok;
    copy_row64h(qh + ((size_t)bh * LSEQ + tok) * DIM, &Ks[j][0]);
  }
  ASYNC_WAIT();
  __syncthreads();
  int wave = tid >> 5, lane = tid & 31;
  int ti = wave;                          // 8 row tiles
  for (int tj = 0; tj < 16; ++tj) {       // 16 col tiles
    v8f acc = {};
#pragma unroll
    for (int ks = 0; ks < 64; ks += 32) {
      v16h a = frag_rowmajor(&Qs[0][0], ti * 16, ks, 72, lane);
      v16h b = frag_rowmajor(&Ks[0][0], tj * 16, ks, 72, lane); // B[k][n]=Ks[n][k]
      acc = __builtin_amdgcn_wmma_f32_16x16x32_f16(false, a, false, b,
                                                   (short)0, acc, false, false);
    }
#pragma unroll
    for (int vv = 0; vv < 8; ++vv) {
      int i = ti * 16 + vv + ((lane >> 4) << 3);
      int j = tj * 16 + (lane & 15);
      float s = acc[vv] * 0.125f;         // 1/sqrt(64)
      if (sh_s[i] != kh_s[j]) s = -1.0e9f;
      if (qi_s[i] <  ki_s[j]) s = -1.0e9f;
      if (qi_s[i] == ki_s[j]) s = -1.0e5f;
      scores_g[(((size_t)bh * LSEQ + qi_s[i]) * RR + r) * KW + j] = s;
    }
  }
}

// ---- 5) per-token dup-key count (histogram) + softmax over 1024 ------------
// thread owns elements e = tid*4 .. tid*4+3 (contiguous, single r per thread).
__global__ __launch_bounds__(256) void softmax_kernel(
    const float* __restrict__ scores_g, const int* __restrict__ orig_idx,
    const int* __restrict__ kidx_g, _Float16* __restrict__ pattn) {
  __shared__ int   cnt[LSEQ];
  __shared__ int   key[RR * KW];
  __shared__ float sv[RR * KW];
  __shared__ float red[256];
  int bid = blockIdx.x;
  int bh = bid >> 11, t = bid & (LSEQ - 1);
  int tid = threadIdx.x;
  for (int e = tid; e < LSEQ; e += 256) cnt[e] = 0;
  const float* srow = scores_g + ((size_t)bh * LSEQ + t) * (RR * KW);
  ((float4*)sv)[tid] = ((const float4*)srow)[tid];   // b128 load of 4 scores
  __syncthreads();
  {
    int r  = tid >> 6;                  // 256 elements per r, 4 per thread
    int j0 = (tid * 4) & 255;
    int p  = orig_idx[(size_t)(bh * RR + r) * LSEQ + t];
    const int* krow = kidx_g + ((size_t)(bh * RR + r) * NB2 + (p >> 7)) * KW;
#pragma unroll
    for (int qq = 0; qq < 4; ++qq) {
      int k = krow[j0 + qq];
      key[tid * 4 + qq] = k;
      atomicAdd(&cnt[k], 1);
    }
  }
  __syncthreads();
  float lmax = -3.4e38f;
#pragma unroll
  for (int qq = 0; qq < 4; ++qq) {
    int e = tid * 4 + qq;
    float s = sv[e] - logf((float)cnt[key[e]]);
    sv[e] = s;
    lmax = fmaxf(lmax, s);
  }
  red[tid] = lmax;
  __syncthreads();
  for (int s = 128; s > 0; s >>= 1) { if (tid < s) red[tid] = fmaxf(red[tid], red[tid + s]); __syncthreads(); }
  float m = red[0];
  __syncthreads();
  float lsum = 0.f;
#pragma unroll
  for (int qq = 0; qq < 4; ++qq) {
    int e = tid * 4 + qq;
    float ex = __expf(sv[e] - m);
    sv[e] = ex;
    lsum += ex;
  }
  red[tid] = lsum;
  __syncthreads();
  for (int s = 128; s > 0; s >>= 1) { if (tid < s) red[tid] += red[tid + s]; __syncthreads(); }
  float inv = 1.0f / red[0];
  union { _Float16 h[4]; uint2 u; } pk;
#pragma unroll
  for (int qq = 0; qq < 4; ++qq) pk.h[qq] = (_Float16)(sv[tid * 4 + qq] * inv);
  _Float16* prow = pattn + ((size_t)bh * LSEQ + t) * (RR * KW);
  ((uint2*)prow)[tid] = pk.u;                         // packed 4-half store
}

// ---- 6) zero output --------------------------------------------------------
__global__ void zero_kernel(float* __restrict__ out, int n) {
  int i = blockIdx.x * 256 + threadIdx.x;
  if (i < n) out[i] = 0.f;
}

// ---- 7) attn: O = P*V via WMMA, atomic scatter-sum over r ------------------
// block = (bh, n, r); K=256 split into two 128-key phases to bound LDS.
__global__ __launch_bounds__(256) void attn_kernel(
    const _Float16* __restrict__ pattn, const _Float16* __restrict__ vh,
    const int* __restrict__ hash_idx, const int* __restrict__ kidx_g,
    float* __restrict__ out) {
  __shared__ _Float16 Ps[CH][136];
  __shared__ _Float16 Vs[CH][72];
  __shared__ int qi_s[CH];
  int blk = blockIdx.x;
  int r = blk & 3, n = (blk >> 2) & 15, bh = blk >> 6;
  int tid = threadIdx.x;
  const int* hidx = hash_idx + (size_t)(bh * RR + r) * LSEQ;
  if (tid < CH) qi_s[tid] = hidx[n * CH + tid];
  __syncthreads();
  int wave = tid >> 5, lane = tid & 31;
  int ti = wave;
  v8f acc[4] = {};
  for (int jp = 0; jp < 2; ++jp) {
    {   // probabilities: each thread stages 64 halfs of one half-row
      int i = tid >> 1, c0 = (tid & 1) * 64;
      const _Float16* src =
          pattn + (((size_t)bh * LSEQ + qi_s[i]) * RR + r) * KW + jp * CH + c0;
      if (jp == 0) __builtin_prefetch(src + CH, 0, 0);  // global_prefetch_b8
      copy_row64h(src, &Ps[i][c0]);
    }
    if (tid < CH) {   // values for keys jp*128 .. jp*128+127
      int tok = kidx_g[((size_t)(bh * RR + r) * NB2 + n) * KW + jp * CH + tid];
      copy_row64h(vh + ((size_t)bh * LSEQ + tok) * DIM, &Vs[tid][0]);
    }
    ASYNC_WAIT();
    __syncthreads();
#pragma unroll
    for (int ks = 0; ks < 4; ++ks) {
      v16h a = frag_rowmajor(&Ps[0][0], ti * 16, ks * 32, 136, lane);
#pragma unroll
      for (int tj = 0; tj < 4; ++tj) {
        v16h b = frag_kmajor(&Vs[0][0], tj * 16, ks * 32, 72, lane); // B[k][n]=Vs[k][n]
        acc[tj] = __builtin_amdgcn_wmma_f32_16x16x32_f16(false, a, false, b,
                                                         (short)0, acc[tj], false, false);
      }
    }
    __syncthreads();
  }
#pragma unroll
  for (int tj = 0; tj < 4; ++tj) {
#pragma unroll
    for (int vv = 0; vv < 8; ++vv) {
      int i = ti * 16 + vv + ((lane >> 4) << 3);
      int d = tj * 16 + (lane & 15);
      atomicAdd(&out[((size_t)bh * LSEQ + qi_s[i]) * DIM + d], acc[tj][vv]);
    }
  }
}

// ---------------------------------------------------------------------------
extern "C" void kernel_launch(void* const* d_in, const int* in_sizes, int n_in,
                              void* d_out, int out_size, void* d_ws, size_t ws_size,
                              hipStream_t stream) {
  const float* q  = (const float*)d_in[0];   // (2,8,2048,64)
  const float* v  = (const float*)d_in[1];   // (2,8,2048,64)
  const float* rm = (const float*)d_in[2];   // (16,64,4,16)
  float* out = (float*)d_out;                // (2,8,2048,64) f32

  char* ws = (char*)d_ws;
  size_t o = 0;
  auto take = [&](size_t bytes) -> void* {
    void* p = ws + o;
    o = (o + bytes + 255) & ~(size_t)255;
    return p;
  };
  float*    rmn      = (float*)take((size_t)BHN * 64 * 64 * 4);
  _Float16* qh       = (_Float16*)take((size_t)BHN * LSEQ * DIM * 2);
  _Float16* vh       = (_Float16*)take((size_t)BHN * LSEQ * DIM * 2);
  int*      hashes   = (int*)take((size_t)BHN * RR * LSEQ * 4);
  int*      hash_idx = (int*)take((size_t)BHN * RR * LSEQ * 4);
  int*      sorted_h = (int*)take((size_t)BHN * RR * LSEQ * 4);
  int*      orig_idx = (int*)take((size_t)BHN * RR * LSEQ * 4);
  int*      kidx_g   = (int*)take((size_t)BHN * RR * NB2 * KW * 4);
  float*    scores_g = (float*)take((size_t)BHN * LSEQ * RR * KW * 4);
  _Float16* pattn    = (_Float16*)take((size_t)BHN * LSEQ * RR * KW * 2);
  (void)ws_size; (void)in_sizes; (void)n_in;

  norm_rm_kernel<<<BHN * RR * NB2, 64, 0, stream>>>(rm, rmn);
  norm_q_hash_kernel<<<BHN * LSEQ, 64, 0, stream>>>(q, v, rmn, qh, vh, hashes);
  sort_kernel<<<BHN * RR, 256, 0, stream>>>(hashes, hash_idx, sorted_h, orig_idx);
  scores_kernel<<<BHN * NB2 * RR, 256, 0, stream>>>(qh, hash_idx, sorted_h, kidx_g, scores_g);
  softmax_kernel<<<BHN * LSEQ, 256, 0, stream>>>(scores_g, orig_idx, kidx_g, pattn);
  zero_kernel<<<(out_size + 255) / 256, 256, 0, stream>>>(out, out_size);
  attn_kernel<<<BHN * NB2 * RR, 256, 0, stream>>>(pattn, vh, hash_idx, kidx_g, out);
}
